// SConvLTI_10058813407858
// MI455X (gfx1250) — matrix operand order repaired and checked by reading, not compiled, not run on hardware
//
#include <hip/hip_runtime.h>
#include <stdint.h>

// ---------------------------------------------------------------------------
// SConv-LTI block model forward for MI455X (gfx1250), wave32 + WMMA bf16.
//   B=2, L=2048, D=1024, DEPTH=4, FF=4096, VOCAB=32000, M = B*L = 4096
// FFT causal conv replaced by exact per-channel complex recurrence:
//   h[t] = r*h[t-1] + z[t],  h[-1] = lc,  r = exp(1e-5 + i*0.001^(d/(D-1)))
// Weights are converted f32 -> bf16 *pre-transposed* once per layer, so both
// GEMM operand tiles (A rows and W^T rows) are (rows x 32 bf16, stride K)
// and are staged into padded LDS rows by the Tensor Data Mover with double
// buffering: tensor_load_to_lds of tile k+1 overlaps WMMA of tile k.
// GEMMs: v_wmma_f32_16x16x32_bf16, f32 accumulate.
// ---------------------------------------------------------------------------

#define DI static __device__ __forceinline__

typedef unsigned short ushort_t;
typedef unsigned int uint_t;
typedef __attribute__((ext_vector_type(16))) __bf16 v16bf;
typedef __attribute__((ext_vector_type(8)))  float  v8f;

union FragAB { v16bf v; uint4 u[2]; };

// Branchless float->bf16 RNE (all values here are finite).
DI ushort_t f2bf(float f) {
  union { float f; unsigned u; } c; c.f = f;
  unsigned u = c.u + (0x7fffu + ((c.u >> 16) & 1u));
  return (ushort_t)(u >> 16);
}

DI float silu_f(float q) { return q / (1.0f + expf(-q)); }

DI float block_reduce_sum(float v, float* red) {
  int tid = threadIdx.x;
  red[tid] = v; __syncthreads();
  for (int s = 128; s > 0; s >>= 1) {
    if (tid < s) red[tid] += red[tid + s];
    __syncthreads();
  }
  float r = red[0];
  __syncthreads();
  return r;
}

// ---------------------------------------------------------------------------
// Tensor Data Mover: 2D tile load (tile_rows x 32 bf16, row stride K elems)
// global -> LDS with hardware padding of 4 DWORDs (16 B) after every 16
// DWORDs (one 64 B row) => LDS row stride 40 elements == BKP.
// D# per CDNA5 ISA ch.8 (group0 128b, group1 256b; groups 2/3 zero for 2D).
// Issued by one wave (EXEC ignored); completion via S_WAIT_TENSORCNT.
// ---------------------------------------------------------------------------
#if __has_builtin(__builtin_amdgcn_tensor_load_to_lds)
#define HAVE_TDM 1
typedef __attribute__((ext_vector_type(4))) unsigned tdm_v4u;
typedef __attribute__((ext_vector_type(8))) int      tdm_v8i;
typedef __attribute__((ext_vector_type(4))) int      tdm_v4i;

DI void tdm_load_tile_2d(unsigned lds_byte_addr, const ushort_t* gsrc,
                         int k_stride_elems, int tile_rows) {
  unsigned long long ga = (unsigned long long)(uintptr_t)gsrc;
  unsigned K = (unsigned)k_stride_elems;
  unsigned R = (unsigned)tile_rows;

  tdm_v4u g0;
  g0[0] = 1u;                                        // count=1 (valid user D#)
  g0[1] = lds_byte_addr;                             // lds_addr
  g0[2] = (unsigned)(ga & 0xffffffffu);              // global_addr[31:0]
  g0[3] = (unsigned)((ga >> 32) & 0x01ffffffu)       // global_addr[56:32]
        | (2u << 30);                                // type=2 ("image")

  tdm_v8i g1;
  g1[0] = (int)((1u << 16)     // data_size = 2 bytes
              | (1u << 20)     // pad_enable
              | (3u << 22)     // pad_interval: 16 DWORDs (one 64 B row)
              | (3u << 25));   // pad_amount: 4 DWORDs (16 B) -> 80 B stride
  g1[1] = (int)((K & 0xffffu) << 16);                // tensor_dim0 lo
  g1[2] = (int)((K >> 16) | ((R & 0xffffu) << 16));  // dim0 hi | tensor_dim1 lo
  g1[3] = (int)(((R >> 16) & 0xffffu) | (32u << 16));// dim1 hi | tile_dim0=32
  g1[4] = (int)(R & 0xffffu);                        // tile_dim1 | tile_dim2=0
  g1[5] = (int)K;                                    // tensor_dim0_stride lo
  g1[6] = 0;                                         // stride hi | dim1_stride lo
  g1[7] = 0;                                         // dim1_stride hi

  tdm_v4i gz = {};
#if defined(__clang_major__) && (__clang_major__ >= 23)
  tdm_v8i gz8 = {};
  __builtin_amdgcn_tensor_load_to_lds(g0, g1, gz, gz, gz8, 0);
#else
  __builtin_amdgcn_tensor_load_to_lds(g0, g1, gz, gz, 0);
#endif
}
#endif

// ---------------------------------------------------------------------------
// One-time per layer: W[K][N] f32 -> WT[N][K] bf16 (32x32 LDS tile transpose).
// Removes all conversion/transpose VALU + scattered b16 stores from the GEMM
// hot loop (previously redone by every M-block touching the same W tile).
// ---------------------------------------------------------------------------
__global__ __launch_bounds__(256) void transpose_f32_bf16T_kernel(
    const float* __restrict__ W, ushort_t* __restrict__ WT, int K, int N) {
  __shared__ ushort_t tile[32][33];
  const int kb = blockIdx.y * 32, nb = blockIdx.x * 32;
  const int tn = threadIdx.x & 31, t8 = threadIdx.x >> 5;  // 0..31, 0..7
#pragma unroll
  for (int i = 0; i < 4; ++i) {
    int k = t8 + i * 8;
    tile[k][tn] = f2bf(W[(size_t)(kb + k) * N + nb + tn]);  // coalesced read
  }
  __syncthreads();
#pragma unroll
  for (int i = 0; i < 4; ++i) {
    int n = t8 + i * 8;
    WT[(size_t)(nb + n) * K + kb + tn] = tile[tn][n];       // coalesced write
  }
}

// ---------------------------------------------------------------------------
// Embedding gather + max_norm=1 renorm.
// ---------------------------------------------------------------------------
__global__ __launch_bounds__(256) void embed_renorm_kernel(
    const int* __restrict__ tokens, const float* __restrict__ emb,
    float* __restrict__ x, int D) {
  __shared__ float red[256];
  int row = blockIdx.x;
  int tok = tokens[row];
  const float* e = emb + (size_t)tok * D;
  float s = 0.f;
  for (int i = threadIdx.x; i < D; i += 256) { float v = e[i]; s += v * v; }
  s = block_reduce_sum(s, red);
  float nrm = sqrtf(s);
  float scale = fminf(1.0f, 1.0f / fmaxf(nrm, 1e-7f));
  for (int i = threadIdx.x; i < D; i += 256)
    x[(size_t)row * D + i] = e[i] * scale;
}

// ---------------------------------------------------------------------------
// RMSNorm -> bf16 (GEMM A-operand producer).
// ---------------------------------------------------------------------------
__global__ __launch_bounds__(256) void rms_bf16_kernel(
    const float* __restrict__ x, const float* __restrict__ g,
    ushort_t* __restrict__ out, int D) {
  __shared__ float red[256];
  int row = blockIdx.x;
  const float* xr = x + (size_t)row * D;
  float s = 0.f;
  for (int i = threadIdx.x; i < D; i += 256) { float v = xr[i]; s += v * v; }
  s = block_reduce_sum(s, red);
  float rs = rsqrtf(s / (float)D + 1e-6f);
  for (int i = threadIdx.x; i < D; i += 256)
    out[(size_t)row * D + i] = f2bf(xr[i] * rs * g[i]);
}

// ---------------------------------------------------------------------------
// Final LayerNorm -> bf16.
// ---------------------------------------------------------------------------
__global__ __launch_bounds__(256) void ln_bf16_kernel(
    const float* __restrict__ x, const float* __restrict__ w,
    const float* __restrict__ b, ushort_t* __restrict__ out, int D) {
  __shared__ float red[256];
  int row = blockIdx.x;
  const float* xr = x + (size_t)row * D;
  float s = 0.f;
  for (int i = threadIdx.x; i < D; i += 256) s += xr[i];
  s = block_reduce_sum(s, red);
  float mu = s / (float)D;
  float vv = 0.f;
  for (int i = threadIdx.x; i < D; i += 256) { float d = xr[i] - mu; vv += d * d; }
  vv = block_reduce_sum(vv, red);
  float rs = rsqrtf(vv / (float)D + 1e-5f);
  for (int i = threadIdx.x; i < D; i += 256)
    out[(size_t)row * D + i] = f2bf((xr[i] - mu) * rs * w[i] + b[i]);
}

// ---------------------------------------------------------------------------
// Complex first-order scan (replaces FFT conv), in-place: z -> h.real.
// ---------------------------------------------------------------------------
__global__ __launch_bounds__(256) void scan_kernel(
    float* __restrict__ z, const float* __restrict__ lc_re,
    const float* __restrict__ lc_im, int Bc, int L, int D) {
  int tid = blockIdx.x * 256 + threadIdx.x;
  if (tid >= Bc * D) return;
  int b = tid / D, d = tid % D;
  float ang = powf(0.001f, (float)d / (float)(D - 1));
  float mag = expf(1e-5f);
  float rre = mag * cosf(ang);
  float rim = mag * sinf(ang);
  float sre = lc_re[d], sim = lc_im[d];   // h[-1] = lc  (carries lc*r^{t+1})
  float* zp = z + ((size_t)b * L) * D + d;
  for (int t = 0; t < L; ++t) {
    float nre = rre * sre - rim * sim + zp[(size_t)t * D];
    float nim = rre * sim + rim * sre;
    sre = nre; sim = nim;
    zp[(size_t)t * D] = sre;              // keep real part only
  }
}

// ---------------------------------------------------------------------------
// WMMA GEMM: block = 256 thr (8 waves), tile 64(M) x 128(N), BK = 32.
// Wave w: rows (w&3)*16..+15, cols (w>>2)*64..+63  -> 4 accum tiles / GEMM.
// A and W^T tiles both (rows x 32 bf16, stride K): TDM-staged into padded
// LDS (BKP=40, 80 B rows), double buffered across K-steps.
// Fragment layouts per CDNA5 ISA 7.12.2:
//   A 16x32 bf16: lane l -> row l&15; K {8h..8h+7, 8h+16..8h+23}, h = l>>4
//   B 32x16 bf16: lane l -> col l&15; K {8h..8h+7, 8h+16..8h+23}  (rows of W^T)
//   C 16x16 f32 : lane l, vgpr v -> (m = v + 8h, n = l&15)
// ---------------------------------------------------------------------------
#define BKP 40

// MODE: 0 = store f32, 1 = store bf16, 2 = accumulate into f32 (residual)
// SAMEA: A1 == A2 -> single A tile staged/loaded, gate reuses it.
template <int MODE, bool SAMEA>
__global__ __launch_bounds__(256) void gemm_dual_gated_kernel(
    const ushort_t* __restrict__ A1, const ushort_t* __restrict__ WT1,
    const float* __restrict__ b1,
    const ushort_t* __restrict__ A2, const ushort_t* __restrict__ WT2,
    const float* __restrict__ b2,
    void* __restrict__ outv, int M, int N, int K) {
  constexpr int NT = SAMEA ? 3 : 4;          // TDM loads per buffer
  __shared__ ushort_t sA1[2][64 * BKP];
  __shared__ ushort_t sA2[SAMEA ? 1 : 2][64 * BKP];
  __shared__ ushort_t sW1[2][128 * BKP];
  __shared__ ushort_t sW2[2][128 * BKP];

  const int tid  = threadIdx.x;
  const int lane = tid & 31;
  const int wv   = tid >> 5;
  const int m0w  = (wv & 3) * 16;
  const int n0w  = (wv >> 2) * 64;
  const int mBase = blockIdx.y * 64;
  const int nBase = blockIdx.x * 128;

  v8f zero = {};
  v8f acc1[4], acc2[4];
#pragma unroll
  for (int j = 0; j < 4; ++j) { acc1[j] = zero; acc2[j] = zero; }

  const int h    = lane >> 4;
  const int aoff = (m0w + (lane & 15)) * BKP + h * 8;

  const ushort_t* gA  = A1  + (size_t)mBase * K;
  const ushort_t* gA2 = A2  + (size_t)mBase * K;
  const ushort_t* gW1 = WT1 + (size_t)nBase * K;
  const ushort_t* gW2 = WT2 + (size_t)nBase * K;
  const int nk = K / 32;

#if defined(HAVE_TDM)
  auto issue = [&](int buf, int kk0) {
    tdm_load_tile_2d((unsigned)(uintptr_t)&sA1[buf][0], gA + kk0, K, 64);
    if constexpr (!SAMEA)
      tdm_load_tile_2d((unsigned)(uintptr_t)&sA2[buf][0], gA2 + kk0, K, 64);
    tdm_load_tile_2d((unsigned)(uintptr_t)&sW1[buf][0], gW1 + kk0, K, 128);
    tdm_load_tile_2d((unsigned)(uintptr_t)&sW2[buf][0], gW2 + kk0, K, 128);
  };
  if (wv == 0) issue(0, 0);
#endif

  for (int ks = 0; ks < nk; ++ks) {
#if defined(HAVE_TDM)
    const int cur = ks & 1;
    if (ks + 1 < nk) {                       // prefetch next tile, keep NT
      if (wv == 0) issue(cur ^ 1, (ks + 1) * 32);
      __builtin_amdgcn_s_wait_tensorcnt(NT); // current buffer's NT done
    } else {
      __builtin_amdgcn_s_wait_tensorcnt(0);
    }
    __syncthreads();
#else
    const int cur = 0;
    {   // manual staging fallback (bf16 copies, no conversion needed)
      const int kk0 = ks * 32;
      const int ar = tid >> 2, ac = (tid & 3) * 8;
      const uint4 va = *(const uint4*)(gA + (size_t)ar * K + kk0 + ac);
      uint_t* d1 = (uint_t*)&sA1[0][ar * BKP + ac];
      d1[0] = va.x; d1[1] = va.y; d1[2] = va.z; d1[3] = va.w;
      if constexpr (!SAMEA) {
        const uint4 vb = *(const uint4*)(gA2 + (size_t)ar * K + kk0 + ac);
        uint_t* d2 = (uint_t*)&sA2[0][ar * BKP + ac];
        d2[0] = vb.x; d2[1] = vb.y; d2[2] = vb.z; d2[3] = vb.w;
      }
      const int wr = tid >> 1, wc = (tid & 1) * 16;
      const uint4 w1a = *(const uint4*)(gW1 + (size_t)wr * K + kk0 + wc);
      const uint4 w1b = *(const uint4*)(gW1 + (size_t)wr * K + kk0 + wc + 8);
      const uint4 w2a = *(const uint4*)(gW2 + (size_t)wr * K + kk0 + wc);
      const uint4 w2b = *(const uint4*)(gW2 + (size_t)wr * K + kk0 + wc + 8);
      uint_t* e1 = (uint_t*)&sW1[0][wr * BKP + wc];
      uint_t* e2 = (uint_t*)&sW2[0][wr * BKP + wc];
      e1[0]=w1a.x; e1[1]=w1a.y; e1[2]=w1a.z; e1[3]=w1a.w;
      e1[4]=w1b.x; e1[5]=w1b.y; e1[6]=w1b.z; e1[7]=w1b.w;
      e2[0]=w2a.x; e2[1]=w2a.y; e2[2]=w2a.z; e2[3]=w2a.w;
      e2[4]=w2b.x; e2[5]=w2b.y; e2[6]=w2b.z; e2[7]=w2b.w;
    }
    __syncthreads();
#endif

    FragAB a1, a2;
    a1.u[0] = *(const uint4*)&sA1[cur][aoff];
    a1.u[1] = *(const uint4*)&sA1[cur][aoff + 16];
    if constexpr (!SAMEA) {
      a2.u[0] = *(const uint4*)&sA2[cur][aoff];
      a2.u[1] = *(const uint4*)&sA2[cur][aoff + 16];
    }
#pragma unroll
    for (int j = 0; j < 4; ++j) {
      const int boff = (n0w + 16 * j + (lane & 15)) * BKP + h * 8;
      FragAB f1, f2;
      f1.u[0] = *(const uint4*)&sW1[cur][boff];
      f1.u[1] = *(const uint4*)&sW1[cur][boff + 16];
      f2.u[0] = *(const uint4*)&sW2[cur][boff];
      f2.u[1] = *(const uint4*)&sW2[cur][boff + 16];
      acc1[j] = __builtin_amdgcn_wmma_f32_16x16x32_bf16(
          false, a1.v, false, f1.v, (short)0, acc1[j], false, false);
      acc2[j] = __builtin_amdgcn_wmma_f32_16x16x32_bf16(
          false, SAMEA ? a1.v : a2.v, false, f2.v, (short)0, acc2[j],
          false, false);
    }
    __syncthreads();
  }

  // ---- epilogue: bias + SiLU gate, then store / bf16-store / residual-add
#pragma unroll
  for (int j = 0; j < 4; ++j) {
    const int n = nBase + n0w + 16 * j + (lane & 15);
    const float bb1 = b1[n], bb2 = b2[n];
#pragma unroll
    for (int v = 0; v < 8; ++v) {
      const int m = mBase + m0w + h * 8 + v;
      float p = acc1[j][v] + bb1;
      float q = acc2[j][v] + bb2;
      float val = p * silu_f(q);
      size_t idx = (size_t)m * N + n;
      if constexpr (MODE == 0)       ((float*)outv)[idx] = val;
      else if constexpr (MODE == 1)  ((ushort_t*)outv)[idx] = f2bf(val);
      else                           ((float*)outv)[idx] += val;
    }
  }
}

// MODE: 0 = store f32 (with bias), 2 = accumulate into f32 (residual + bias)
template <int MODE>
__global__ __launch_bounds__(256) void gemm_single_kernel(
    const ushort_t* __restrict__ A, const ushort_t* __restrict__ WT,
    const float* __restrict__ bias, void* __restrict__ outv,
    int M, int N, int K) {
  __shared__ ushort_t sA[2][64 * BKP];
  __shared__ ushort_t sW[2][128 * BKP];

  const int tid  = threadIdx.x;
  const int lane = tid & 31;
  const int wv   = tid >> 5;
  const int m0w  = (wv & 3) * 16;
  const int n0w  = (wv >> 2) * 64;
  const int mBase = blockIdx.y * 64;
  const int nBase = blockIdx.x * 128;

  v8f zero = {};
  v8f acc[4];
#pragma unroll
  for (int j = 0; j < 4; ++j) acc[j] = zero;

  const int h    = lane >> 4;
  const int aoff = (m0w + (lane & 15)) * BKP + h * 8;

  const ushort_t* gA = A  + (size_t)mBase * K;
  const ushort_t* gW = WT + (size_t)nBase * K;
  const int nk = K / 32;

#if defined(HAVE_TDM)
  auto issue = [&](int buf, int kk0) {
    tdm_load_tile_2d((unsigned)(uintptr_t)&sA[buf][0], gA + kk0, K, 64);
    tdm_load_tile_2d((unsigned)(uintptr_t)&sW[buf][0], gW + kk0, K, 128);
  };
  if (wv == 0) issue(0, 0);
#endif

  for (int ks = 0; ks < nk; ++ks) {
#if defined(HAVE_TDM)
    const int cur = ks & 1;
    if (ks + 1 < nk) {
      if (wv == 0) issue(cur ^ 1, (ks + 1) * 32);
      __builtin_amdgcn_s_wait_tensorcnt(2);
    } else {
      __builtin_amdgcn_s_wait_tensorcnt(0);
    }
    __syncthreads();
#else
    const int cur = 0;
    {
      const int kk0 = ks * 32;
      const int ar = tid >> 2, ac = (tid & 3) * 8;
      const uint4 va = *(const uint4*)(gA + (size_t)ar * K + kk0 + ac);
      uint_t* d1 = (uint_t*)&sA[0][ar * BKP + ac];
      d1[0] = va.x; d1[1] = va.y; d1[2] = va.z; d1[3] = va.w;
      const int wr = tid >> 1, wc = (tid & 1) * 16;
      const uint4 w1a = *(const uint4*)(gW + (size_t)wr * K + kk0 + wc);
      const uint4 w1b = *(const uint4*)(gW + (size_t)wr * K + kk0 + wc + 8);
      uint_t* e1 = (uint_t*)&sW[0][wr * BKP + wc];
      e1[0]=w1a.x; e1[1]=w1a.y; e1[2]=w1a.z; e1[3]=w1a.w;
      e1[4]=w1b.x; e1[5]=w1b.y; e1[6]=w1b.z; e1[7]=w1b.w;
    }
    __syncthreads();
#endif

    FragAB a;
    a.u[0] = *(const uint4*)&sA[cur][aoff];
    a.u[1] = *(const uint4*)&sA[cur][aoff + 16];
#pragma unroll
    for (int j = 0; j < 4; ++j) {
      const int boff = (n0w + 16 * j + (lane & 15)) * BKP + h * 8;
      FragAB f;
      f.u[0] = *(const uint4*)&sW[cur][boff];
      f.u[1] = *(const uint4*)&sW[cur][boff + 16];
      acc[j] = __builtin_amdgcn_wmma_f32_16x16x32_bf16(
          false, a.v, false, f.v, (short)0, acc[j], false, false);
    }
    __syncthreads();
  }

#pragma unroll
  for (int j = 0; j < 4; ++j) {
    const int n = nBase + n0w + 16 * j + (lane & 15);
    const float bb = bias[n];
#pragma unroll
    for (int v = 0; v < 8; ++v) {
      const int m = mBase + m0w + h * 8 + v;
      float val = acc[j][v] + bb;
      size_t idx = (size_t)m * N + n;
      if constexpr (MODE == 0) ((float*)outv)[idx] = val;
      else                     ((float*)outv)[idx] += val;
    }
  }
}

// ---------------------------------------------------------------------------
// Host orchestration.
// ---------------------------------------------------------------------------
extern "C" void kernel_launch(void* const* d_in, const int* in_sizes, int n_in,
                              void* d_out, int out_size, void* d_ws, size_t ws_size,
                              hipStream_t stream) {
  (void)in_sizes; (void)n_in; (void)out_size; (void)ws_size;
  constexpr int Bc = 2, L = 2048, D = 1024, DEPTH = 4, FF = 4096, VOCAB = 32000;
  constexpr int M = Bc * L;  // 4096 token rows

  const int*   tokens = (const int*)  d_in[0];
  const float* emb    = (const float*)d_in[1];
  const float* Wz     = (const float*)d_in[2];
  const float* bz     = (const float*)d_in[3];
  const float* Wza    = (const float*)d_in[4];
  const float* bza    = (const float*)d_in[5];
  const float* Wy     = (const float*)d_in[6];
  const float* by     = (const float*)d_in[7];
  const float* Wya    = (const float*)d_in[8];
  const float* bya    = (const float*)d_in[9];
  const float* lc_re  = (const float*)d_in[10];
  const float* lc_im  = (const float*)d_in[11];
  const float* g_in   = (const float*)d_in[12];
  const float* g_sc   = (const float*)d_in[13];
  const float* g_ffn  = (const float*)d_in[14];
  const float* Wf     = (const float*)d_in[15];
  const float* bfb    = (const float*)d_in[16];
  const float* Wfa    = (const float*)d_in[17];
  const float* bfa    = (const float*)d_in[18];
  const float* Wfo    = (const float*)d_in[19];
  const float* bfo    = (const float*)d_in[20];
  const float* ln_w   = (const float*)d_in[21];
  const float* ln_b   = (const float*)d_in[22];
  const float* Wout   = (const float*)d_in[23];
  const float* bout   = (const float*)d_in[24];

  // Workspace carve (~144 MB): activations + per-layer transposed bf16 weights
  char* w = (char*)d_ws;
  float*    x   = (float*)w;     w += (size_t)M * D * sizeof(float);
  float*    z   = (float*)w;     w += (size_t)M * D * sizeof(float);
  ushort_t* xn  = (ushort_t*)w;  w += (size_t)M * D * sizeof(ushort_t);
  ushort_t* hn  = (ushort_t*)w;  w += (size_t)M * D * sizeof(ushort_t);
  ushort_t* gg  = (ushort_t*)w;  w += (size_t)M * FF * sizeof(ushort_t);
  ushort_t* wt  = (ushort_t*)w;  // 64 MB region, reused per layer / for Wout
  constexpr size_t DD = (size_t)D * D, DF = (size_t)D * FF;
  ushort_t* wtz  = wt;
  ushort_t* wtza = wt + DD;
  ushort_t* wty  = wt + 2 * DD;
  ushort_t* wtya = wt + 3 * DD;
  ushort_t* wtf  = wt + 4 * DD;
  ushort_t* wtfa = wt + 4 * DD + DF;
  ushort_t* wtfo = wt + 4 * DD + 2 * DF;
  ushort_t* wtout = wt;          // final projection reuses the region

  dim3 blk(256);
  dim3 gD(D / 128, M / 64);       // N=1024 GEMMs
  dim3 gF(FF / 128, M / 64);      // N=4096 GEMMs
  dim3 gV(VOCAB / 128, M / 64);   // N=32000 GEMM

  embed_renorm_kernel<<<M, blk, 0, stream>>>(tokens, emb, x, D);

  for (int i = 0; i < DEPTH; ++i) {
    const size_t oDD = (size_t)i * D * D;
    const size_t oDF = (size_t)i * D * FF;
    const size_t oFD = (size_t)i * FF * D;
    const size_t oD  = (size_t)i * D;
    const size_t oF  = (size_t)i * FF;

    // One-time conversion+transpose of this layer's weights to bf16 [N][K]
    transpose_f32_bf16T_kernel<<<dim3(D/32,  D/32),  blk, 0, stream>>>(Wz  + oDD, wtz,  D,  D);
    transpose_f32_bf16T_kernel<<<dim3(D/32,  D/32),  blk, 0, stream>>>(Wza + oDD, wtza, D,  D);
    transpose_f32_bf16T_kernel<<<dim3(D/32,  D/32),  blk, 0, stream>>>(Wy  + oDD, wty,  D,  D);
    transpose_f32_bf16T_kernel<<<dim3(D/32,  D/32),  blk, 0, stream>>>(Wya + oDD, wtya, D,  D);
    transpose_f32_bf16T_kernel<<<dim3(FF/32, D/32),  blk, 0, stream>>>(Wf  + oDF, wtf,  D,  FF);
    transpose_f32_bf16T_kernel<<<dim3(FF/32, D/32),  blk, 0, stream>>>(Wfa + oDF, wtfa, D,  FF);
    transpose_f32_bf16T_kernel<<<dim3(D/32,  FF/32), blk, 0, stream>>>(Wfo + oFD, wtfo, FF, D);

    // SConv sublayer
    rms_bf16_kernel<<<M, blk, 0, stream>>>(x, g_sc + oD, xn, D);
    gemm_dual_gated_kernel<0, true><<<gD, blk, 0, stream>>>(
        xn, wtz, bz + oD, xn, wtza, bza + oD, (void*)z, M, D, D);
    scan_kernel<<<(Bc * D + 255) / 256, blk, 0, stream>>>(
        z, lc_re + oD, lc_im + oD, Bc, L, D);
    rms_bf16_kernel<<<M, blk, 0, stream>>>(z, g_in + oD, hn, D);
    gemm_dual_gated_kernel<2, false><<<gD, blk, 0, stream>>>(
        hn, wty, by + oD, xn, wtya, bya + oD, (void*)x, M, D, D);

    // FFN SwiGLU sublayer
    rms_bf16_kernel<<<M, blk, 0, stream>>>(x, g_ffn + oD, xn, D);
    gemm_dual_gated_kernel<1, true><<<gF, blk, 0, stream>>>(
        xn, wtf, bfb + oF, xn, wtfa, bfa + oF, (void*)gg, M, FF, D);
    gemm_single_kernel<2><<<gD, blk, 0, stream>>>(
        gg, wtfo, bfo + oD, (void*)x, M, D, FF);
  }

  // Final LayerNorm + vocab projection (f32 out)
  ln_bf16_kernel<<<M, blk, 0, stream>>>(x, ln_w, ln_b, xn, D);
  transpose_f32_bf16T_kernel<<<dim3(VOCAB/32, D/32), blk, 0, stream>>>(
      Wout, wtout, D, VOCAB);
  gemm_single_kernel<0><<<gV, blk, 0, stream>>>(
      xn, wtout, bout, d_out, M, VOCAB, D);
}
